// MultiHeadAttention_30030411333990
// MI455X (gfx1250) — matrix operand order repaired and checked by reading, not compiled, not run on hardware
//
#include <hip/hip_runtime.h>
#include <hip/hip_bf16.h>

// ---------------------------------------------------------------------------
// MHA forward for MI455X (gfx1250), wave32.
//  - f16 WMMA (v_wmma_f32_16x16x32_f16) for all 4 projections + attention
//  - TDM (tensor_load_to_lds) double-buffered K/V tile staging in attention
// ---------------------------------------------------------------------------

typedef __attribute__((ext_vector_type(16))) _Float16 v16h;
typedef __attribute__((ext_vector_type(8)))  _Float16 v8h;
typedef __attribute__((ext_vector_type(4)))  _Float16 v4h;
typedef __attribute__((ext_vector_type(8)))  float    v8f;
typedef __attribute__((ext_vector_type(4))) unsigned int u32x4;
typedef __attribute__((ext_vector_type(8))) int          i32x8;
typedef __attribute__((ext_vector_type(4))) int          i32x4;

#define HID_  1024
#define HEADS_ 16
#define HD_    64
#define B_     4
#define S_     2048
#define MTOT_  (B_ * S_)   // 8192
#define KROW_  72          // LDS tile row stride in halves (64 + 16B TDM pad)

#if defined(__HIP_DEVICE_COMPILE__) && __has_builtin(__builtin_amdgcn_tensor_load_to_lds)
#define ATHENA_TDM 1
#else
#define ATHENA_TDM 0
#endif

// Build a v16h A-fragment from two contiguous 8-half (16B) runs.
static __device__ __forceinline__ v16h ld_pair8(const _Float16* p0, const _Float16* p1) {
  v8h a = *(const v8h*)p0;
  v8h b = *(const v8h*)p1;
  v16h r;
#pragma unroll
  for (int e = 0; e < 8; ++e) { r[e] = a[e]; r[e + 8] = b[e]; }
  return r;
}
// Build a v16h B-fragment from one contiguous 16-half (32B) run.
static __device__ __forceinline__ v16h ld_run16(const _Float16* p) {
  return ld_pair8(p, p + 8);
}
static __device__ __forceinline__ v8f wmma_f16(v16h a, v16h b, v8f c) {
  // (neg_a, A, neg_b, B, c_mod, C, reuse_a, reuse_b)
  return __builtin_amdgcn_wmma_f32_16x16x32_f16(false, a, false, b, (short)0, c,
                                                false, false);
}
static __device__ __forceinline__ v8f vzero8() {
  v8f z;
#pragma unroll
  for (int e = 0; e < 8; ++e) z[e] = 0.0f;
  return z;
}

// ---------------------------------------------------------------------------
// TDM: 2D f16 tile DMA (global -> LDS), per CDNA5 ISA 08_async_tensor §8.
// LDS rows padded 16B per 128B row (pad_interval=4 -> 32 DWORDs,
// pad_amount=3 -> 4 DWORDs) => LDS row stride = KROW_ halves, bank-conflict
// free for 16-row-strided fragment reads. Tracked by TENSORcnt.
// ---------------------------------------------------------------------------
static __device__ __forceinline__ void tdm_load_tile_f16(
    void* lds_dst, const _Float16* gsrc, int tile_w, int tile_h,
    unsigned long long row_stride_elems, unsigned tensor_w, unsigned tensor_h) {
#if ATHENA_TDM
  const unsigned long long ga = (unsigned long long)(uintptr_t)gsrc;
  u32x4 g0;
  g0[0] = 1u;                                   // count=1 (valid, user mode)
  // Flat LDS aperture keeps the AS3 byte offset in addr[31:0].
  g0[1] = (unsigned)(uintptr_t)lds_dst;         // lds_addr (bytes)
  g0[2] = (unsigned)(ga & 0xffffffffull);       // global_addr[31:0]
  g0[3] = (unsigned)((ga >> 32) & 0x1ffffffull) // global_addr[56:32]
          | (2u << 30);                         // type=2 ("image")

  const unsigned long long flags =
      (1ull << 16)    // data_size = 1 -> 2-byte elements
    | (1ull << 20)    // pad_enable
    | (4ull << 22)    // pad_interval: 32 DWORDs (one 128B row)
    | (3ull << 25);   // pad_amount:   4 DWORDs (16B)
  const unsigned long long q0 =
      flags | ((unsigned long long)(tensor_w & 0xffffu) << 48);
  const unsigned long long q1 =
      (unsigned long long)(tensor_w >> 16)
    | ((unsigned long long)tensor_h << 16)           // tensor_dim1 [111:80]
    | ((unsigned long long)(tile_w & 0xffff) << 48); // tile_dim0   [127:112]
  const unsigned long long q2 =
      (unsigned long long)(tile_h & 0xffff)          // tile_dim1   [143:128]
    | ((row_stride_elems & 0xffffffffull) << 32);    // stride[31:0][191:160]
  const unsigned long long q3 =
      (row_stride_elems >> 32) & 0xffffull;          // stride[47:32][207:192]
  i32x8 g1;
  g1[0] = (int)q0; g1[1] = (int)(q0 >> 32);
  g1[2] = (int)q1; g1[3] = (int)(q1 >> 32);
  g1[4] = (int)q2; g1[5] = (int)(q2 >> 32);
  g1[6] = (int)q3; g1[7] = (int)(q3 >> 32);
  const i32x4 gz = {0, 0, 0, 0};                // groups 2/3: 2-D tile
#if defined(__clang_major__) && (__clang_major__ >= 23)
  // amdgpu-toolchain (clang-23+): 6-arg form with extra int32x8 before cpol.
  const i32x8 gz8 = {0, 0, 0, 0, 0, 0, 0, 0};
  __builtin_amdgcn_tensor_load_to_lds(g0, g1, gz, gz, gz8, 0);
#else
  // ROCm 7.2 (clang-22): 5-arg form.
  __builtin_amdgcn_tensor_load_to_lds(g0, g1, gz, gz, 0);
#endif
#else
  (void)lds_dst; (void)gsrc; (void)tile_w; (void)tile_h;
  (void)row_stride_elems; (void)tensor_w; (void)tensor_h;
#endif
}

// ---------------------------------------------------------------------------
// fp32 -> f16 conversion (vectorized x4)
// ---------------------------------------------------------------------------
__global__ __launch_bounds__(256) void cvt_f32_f16_kernel(
    const float* __restrict__ src, _Float16* __restrict__ dst, int n4) {
  int i = blockIdx.x * blockDim.x + threadIdx.x;
  if (i >= n4) return;
  const float4 v = ((const float4*)src)[i];
  v4h h;
  h[0] = (_Float16)v.x; h[1] = (_Float16)v.y;
  h[2] = (_Float16)v.z; h[3] = (_Float16)v.w;
  ((v4h*)dst)[i] = h;
}

// ---------------------------------------------------------------------------
// Generic WMMA GEMM:  C[i,j] = sum_k RA[i,k] * RB[j,k]   (both k-contiguous)
// Wave computes a 16(i) x 64(j) tile; 8 waves/block => 32(i) x 256(j).
// MODE 0: out f16, normal layout [j_seq][HID], feature = i; val=(acc+bias)*scale
// MODE 1: out f16, transposed   [j_feat][MTOT], seq = i;  val= acc+bias
// MODE 2: out f32, normal layout [j_seq][HID], feature = i (final projection)
// ---------------------------------------------------------------------------
template <int MODE>
__global__ __launch_bounds__(256) void gemm_wmma_kernel(
    const _Float16* __restrict__ RA, const _Float16* __restrict__ RB,
    const float* __restrict__ bias, void* __restrict__ outp, float scale) {
  const int lane = threadIdx.x & 31;
  const int wave = threadIdx.x >> 5;
  const int ll = lane & 15, lh = lane >> 4;
  const int i0 = blockIdx.y * 32 + (wave >> 2) * 16;
  const int j0 = blockIdx.x * 256 + (wave & 3) * 64;

  const v8f z = vzero8();
  v8f acc[4] = {z, z, z, z};

  const _Float16* arow = RA + (size_t)(i0 + ll) * HID_;
#pragma unroll 2
  for (int kk = 0; kk < HID_; kk += 32) {
    __builtin_prefetch(arow + kk + 64, 0, 0);  // global_prefetch_b8
    v16h a = ld_run16(arow + kk + lh * 16);
#pragma unroll
    for (int c = 0; c < 4; ++c) {
      const _Float16* brow =
          RB + (size_t)(j0 + c * 16 + ll) * HID_ + kk + lh * 16;
      v16h b = ld_run16(brow);
      acc[c] = wmma_f16(a, b, acc[c]);
    }
  }

  // C layout: lane = column j (N), VGPR r = row i (M = r + 8*lh).
  if (MODE == 0) {
    _Float16* out = (_Float16*)outp;
    const int fb = i0 + lh * 8;
    float bv[8];
#pragma unroll
    for (int r = 0; r < 8; ++r) bv[r] = bias[fb + r];
#pragma unroll
    for (int c = 0; c < 4; ++c) {
      const int s = j0 + c * 16 + ll;
      v8h h;
#pragma unroll
      for (int r = 0; r < 8; ++r)
        h[r] = (_Float16)((acc[c][r] + bv[r]) * scale);
      *(v8h*)(out + (size_t)s * HID_ + fb) = h;
    }
  } else if (MODE == 2) {
    float* out = (float*)outp;
    const int fb = i0 + lh * 8;
    float bv[8];
#pragma unroll
    for (int r = 0; r < 8; ++r) bv[r] = bias[fb + r];
#pragma unroll
    for (int c = 0; c < 4; ++c) {
      const int s = j0 + c * 16 + ll;
      float* po = out + (size_t)s * HID_ + fb;
      float4 t0, t1;
      t0.x = acc[c][0] + bv[0]; t0.y = acc[c][1] + bv[1];
      t0.z = acc[c][2] + bv[2]; t0.w = acc[c][3] + bv[3];
      t1.x = acc[c][4] + bv[4]; t1.y = acc[c][5] + bv[5];
      t1.z = acc[c][6] + bv[6]; t1.w = acc[c][7] + bv[7];
      *(float4*)po = t0;
      *(float4*)(po + 4) = t1;
    }
  } else {  // MODE 1: V^T output [feature][MTOT]
    _Float16* out = (_Float16*)outp;
#pragma unroll
    for (int c = 0; c < 4; ++c) {
      const int f = j0 + c * 16 + ll;
      const float bvv = bias[f];
      v8h h;
#pragma unroll
      for (int r = 0; r < 8; ++r) h[r] = (_Float16)(acc[c][r] + bvv);
      *(v8h*)(out + (size_t)f * MTOT_ + i0 + lh * 8) = h;
    }
  }
}

// ---------------------------------------------------------------------------
// Flash attention. Block = 4 waves sharing one (b,h); each wave owns 16 query
// rows. Wave 0 drives a double-buffered TDM pipeline: 64x64 f16 K-tile and
// V^T-tile are DMA'd into LDS (TENSORcnt + one block barrier per tile) and
// all 4 waves consume them as WMMA fragments from LDS.
// Per 64-key tile:  S^T = K·Q^T (8 WMMA) -> lane-local online softmax
// (lane = query column) -> P^T registers are *already* in the A-fragment
// layout for O += P·V (8 WMMA). Q pre-scaled by 1/8.
// ---------------------------------------------------------------------------
__global__ __launch_bounds__(128) void attn_fa_kernel(
    const _Float16* __restrict__ Q, const _Float16* __restrict__ Kb,
    const _Float16* __restrict__ Vt, const int* __restrict__ mask,
    _Float16* __restrict__ O) {
  __shared__ float smask[S_];
  __shared__ _Float16 kt[2][64 * KROW_];  // [key row][d], padded rows
  __shared__ _Float16 vt[2][64 * KROW_];  // [d row][kk],  padded rows

  const int nqt = S_ / 64;  // 32 query tiles per (b,h)
  const int b = blockIdx.x / (HEADS_ * nqt);
  const int r0 = blockIdx.x % (HEADS_ * nqt);
  const int h = r0 / nqt;
  const int qt = r0 % nqt;

  for (int i = threadIdx.x; i < S_; i += 128)
    smask[i] = mask[b * S_ + i] ? 1.0f : 0.0f;

  const int lane = threadIdx.x & 31;
  const int wave = threadIdx.x >> 5;
  const int ll = lane & 15, lh = lane >> 4;
  const int q0 = qt * 64 + wave * 16;

  const _Float16* kbase = Kb + (size_t)(b * S_) * HID_ + h * HD_;
  const _Float16* vbase = Vt + (size_t)(h * HD_) * MTOT_ + b * S_;

#if ATHENA_TDM
  if (wave == 0) {  // prologue: stage tile 0 into buffer 0
    tdm_load_tile_f16(&kt[0][0], kbase, 64, 64, HID_, HID_, MTOT_);
    tdm_load_tile_f16(&vt[0][0], vbase, 64, 64, MTOT_, MTOT_, HID_);
  }
#endif

  // Q as B-fragment (lane = query column, contiguous d-run), kept resident.
  const _Float16* qrow = Q + (size_t)(b * S_ + q0 + ll) * HID_ + h * HD_;
  const v16h qf0 = ld_run16(qrow + lh * 16);        // d = 0..31
  const v16h qf1 = ld_run16(qrow + 32 + lh * 16);   // d = 32..63

  const v8f z = vzero8();
  v8f o[4] = {z, z, z, z};           // O tile 16q x 64d (lane=d col, vgpr=q row)
  float m_i = -3.0e38f, l_i = 0.0f;  // per-lane state for query q = ll

#if !ATHENA_TDM
  __syncthreads();  // smask ready (TDM path syncs inside the loop)
#endif

  for (int jt = 0; jt < S_ / 64; ++jt) {
    const int j0 = jt * 64;
    const int cur = jt & 1;
#if ATHENA_TDM
    if (wave == 0) __builtin_amdgcn_s_wait_tensorcnt(0);
    __syncthreads();  // tile[cur] visible to all; prior reads of tile[nxt] done
    if (wave == 0 && jt + 1 < S_ / 64) {  // stage next tile into other buffer
      const int jn = j0 + 64;
      tdm_load_tile_f16(&kt[cur ^ 1][0], kbase + (size_t)jn * HID_, 64, 64,
                        HID_, HID_, MTOT_);
      tdm_load_tile_f16(&vt[cur ^ 1][0], vbase + jn, 64, 64, MTOT_, MTOT_,
                        HID_);
    }
#endif

    // --- S^T = K · Q^T : 4 key sub-tiles x 2 d-steps -----------------------
    v8f sc[4] = {z, z, z, z};  // lane = q col, vgpr r = key (sub*16 + r + 8*lh)
#pragma unroll
    for (int sub = 0; sub < 4; ++sub) {
#if ATHENA_TDM
      const _Float16* krow = &kt[cur][(sub * 16 + ll) * KROW_];
#else
      const _Float16* krow =
          Kb + (size_t)(b * S_ + j0 + sub * 16 + ll) * HID_ + h * HD_;
#endif
      v16h ka = ld_run16(krow + lh * 16);
      sc[sub] = wmma_f16(ka, qf0, sc[sub]);
      v16h kc = ld_run16(krow + 32 + lh * 16);
      sc[sub] = wmma_f16(kc, qf1, sc[sub]);
    }

    // --- online softmax (softmax dim is lane-local + one xor-16 shuffle) ---
    float tm = -3.0e38f;
#pragma unroll
    for (int sub = 0; sub < 4; ++sub)
#pragma unroll
      for (int r = 0; r < 8; ++r) tm = fmaxf(tm, sc[sub][r]);
    tm = fmaxf(tm, __shfl_xor(tm, 16, 32));
    const float mnew = fmaxf(m_i, tm);
    const float alpha = __expf(m_i - mnew);  // first iter: exp(-inf)=0

    float p[4][8];
    float rs = 0.0f;
#pragma unroll
    for (int sub = 0; sub < 4; ++sub) {
      const float* mp = &smask[j0 + sub * 16 + lh * 8];
#pragma unroll
      for (int r = 0; r < 8; ++r) {
        // mask as 0/1 multiplier post-exp == -1e10 bias pre-softmax
        const float pv = __expf(sc[sub][r] - mnew) * mp[r];
        p[sub][r] = pv;
        rs += pv;
      }
    }
    rs += __shfl_xor(rs, 16, 32);
    l_i = l_i * alpha + rs;
    m_i = mnew;

    // Rescale O: row q = r + 8*lh; its alpha lives in lane (r + 8*lh).
#pragma unroll
    for (int r = 0; r < 8; ++r) {
      const float ar = __shfl(alpha, r + lh * 8, 32);
#pragma unroll
      for (int sd = 0; sd < 4; ++sd) o[sd][r] *= ar;
    }

    // P^T registers -> A-fragments directly (kk = sub*16 + r + 8*lh).
    v16h pa, pb;
#pragma unroll
    for (int r = 0; r < 8; ++r) {
      pa[r] = (_Float16)p[0][r];
      pa[r + 8] = (_Float16)p[1][r];
      pb[r] = (_Float16)p[2][r];
      pb[r + 8] = (_Float16)p[3][r];
    }

    // --- O += P · V  (V^T rows are kk-contiguous) --------------------------
#pragma unroll
    for (int sd = 0; sd < 4; ++sd) {
#if ATHENA_TDM
      const _Float16* vrow = &vt[cur][(sd * 16 + ll) * KROW_ + lh * 8];
#else
      const _Float16* vrow = Vt + (size_t)(h * HD_ + sd * 16 + ll) * MTOT_ +
                             b * S_ + j0 + lh * 8;
#endif
      o[sd] = wmma_f16(pa, ld_pair8(vrow, vrow + 16), o[sd]);        // kk 0..31
      o[sd] = wmma_f16(pb, ld_pair8(vrow + 32, vrow + 48), o[sd]);   // kk 32..63
    }
  }

  // --- epilogue: divide by l, store normal layout [s][HID] -----------------
  const float linv = 1.0f / fmaxf(l_i, 1e-20f);
#pragma unroll
  for (int r = 0; r < 8; ++r) {
    const float lr = __shfl(linv, r + lh * 8, 32);
    _Float16* orow =
        O + (size_t)(b * S_ + q0 + r + 8 * lh) * HID_ + h * HD_;
#pragma unroll
    for (int sd = 0; sd < 4; ++sd)
      orow[sd * 16 + ll] = (_Float16)(o[sd][r] * lr);
  }
}

// ---------------------------------------------------------------------------
extern "C" void kernel_launch(void* const* d_in, const int* in_sizes, int n_in,
                              void* d_out, int out_size, void* d_ws,
                              size_t ws_size, hipStream_t stream) {
  (void)in_sizes; (void)n_in; (void)out_size; (void)ws_size;
  const float* q32 = (const float*)d_in[0];
  const float* k32 = (const float*)d_in[1];
  const float* v32 = (const float*)d_in[2];
  const int*   msk = (const int*)d_in[3];
  const float* Wq = (const float*)d_in[4];
  const float* bq = (const float*)d_in[5];
  const float* Wk = (const float*)d_in[6];
  const float* bk = (const float*)d_in[7];
  const float* Wv = (const float*)d_in[8];
  const float* bv = (const float*)d_in[9];
  const float* Wo = (const float*)d_in[10];
  const float* bo = (const float*)d_in[11];

  _Float16* w = (_Float16*)d_ws;
  const size_t NX = (size_t)MTOT_ * HID_;  // 8M halves
  const size_t NW = (size_t)HID_ * HID_;   // 1M halves
  _Float16* Xq  = w;            // reused later as attention output
  _Float16* Xk  = Xq + NX;
  _Float16* Xv  = Xk + NX;
  _Float16* WqH = Xv + NX;
  _Float16* WkH = WqH + NW;
  _Float16* WvH = WkH + NW;
  _Float16* WoH = WvH + NW;
  _Float16* Qb  = WoH + NW;     // [MTOT][HID], pre-scaled by 1/8
  _Float16* Kb  = Qb + NX;      // [MTOT][HID]
  _Float16* Vt  = Kb + NX;      // [HID][MTOT] (transposed)
  _Float16* Oatt = Xq;          // alias: Xq is dead after the Q projection

  const int T = 256;
  cvt_f32_f16_kernel<<<(int)(NX / 4 / T), T, 0, stream>>>(q32, Xq, (int)(NX / 4));
  cvt_f32_f16_kernel<<<(int)(NX / 4 / T), T, 0, stream>>>(k32, Xk, (int)(NX / 4));
  cvt_f32_f16_kernel<<<(int)(NX / 4 / T), T, 0, stream>>>(v32, Xv, (int)(NX / 4));
  cvt_f32_f16_kernel<<<(int)(NW / 4 / T), T, 0, stream>>>(Wq, WqH, (int)(NW / 4));
  cvt_f32_f16_kernel<<<(int)(NW / 4 / T), T, 0, stream>>>(Wk, WkH, (int)(NW / 4));
  cvt_f32_f16_kernel<<<(int)(NW / 4 / T), T, 0, stream>>>(Wv, WvH, (int)(NW / 4));
  cvt_f32_f16_kernel<<<(int)(NW / 4 / T), T, 0, stream>>>(Wo, WoH, (int)(NW / 4));

  // Q/K projections: i = features (rows of W), j = seq (rows of X).
  dim3 gP(MTOT_ / 256, HID_ / 32);
  gemm_wmma_kernel<0><<<gP, 256, 0, stream>>>(WqH, Xq, bq, Qb, 0.125f);
  gemm_wmma_kernel<0><<<gP, 256, 0, stream>>>(WkH, Xk, bk, Kb, 1.0f);
  // V projection transposed out: i = seq (rows of X), j = features (rows of W).
  dim3 gV(HID_ / 256, MTOT_ / 32);
  gemm_wmma_kernel<1><<<gV, 256, 0, stream>>>(Xv, WvH, bv, Vt, 1.0f);

  attn_fa_kernel<<<B_ * HEADS_ * (S_ / 64), 128, 0, stream>>>(Qb, Kb, Vt, msk,
                                                              Oatt);

  // Final projection, f32 straight into d_out.
  gemm_wmma_kernel<2><<<gP, 256, 0, stream>>>(WoH, Oatt, bo, d_out, 1.0f);
}
